// fc_connect_1589137899947
// MI455X (gfx1250) — matrix operand order repaired and checked by reading, not compiled, not run on hardware
//
#include <hip/hip_runtime.h>
#include <hip/hip_bf16.h>

// Problem sizes (fixed by the reference)
constexpr int B = 256, R = 100, D = 6168, H = 128, C = 100;
constexpr int KC   = 512;                    // K-chunk staged in LDS
constexpr int ASTR = KC + 8;                 // padded LDS row stride (bf16 elems)
constexpr int NCHUNK = (D + KC - 1) / KC;    // 13, last chunk has 24 valid k
constexpr int RT = (R + 15) / 16;            // 7 row tiles of 16

typedef __attribute__((ext_vector_type(16))) __bf16 v16bf;
typedef __attribute__((ext_vector_type(8)))  __bf16 v8bf;
typedef __attribute__((ext_vector_type(8)))  float  v8f;
typedef __attribute__((ext_vector_type(4)))  int    i4;

// One-time fp32 -> bf16 (RNE) conversion of W1 into workspace.
__global__ void w1_to_bf16(const float* __restrict__ W1,
                           unsigned short* __restrict__ W1b, int n) {
  int i = blockIdx.x * blockDim.x + threadIdx.x;
  if (i < n) {
    unsigned u = __float_as_uint(W1[i]);
    W1b[i] = (unsigned short)((u + 0x7FFFu + ((u >> 16) & 1u)) >> 16);
  }
}

__global__ __launch_bounds__(256) void noisy_mlp(
    const int* __restrict__ x, const int* __restrict__ noise,
    const unsigned short* __restrict__ W1b, const float* __restrict__ b1,
    const float* __restrict__ W2, const float* __restrict__ b2,
    float* __restrict__ out)
{
  __shared__ unsigned short tileA[2][16][ASTR];   // double-buffered bf16 A tile (~33 KB)
  __shared__ float hpart[8][32];
  __shared__ float hsum[H];

  const int b      = blockIdx.x;
  const int tid    = threadIdx.x;
  const int wave   = tid >> 5;          // 8 waves: one 16-col N-tile each
  const int lane   = tid & 31;
  const int laneN  = lane & 15;
  const int laneHi = lane >> 4;
  const int n0     = wave * 16;

  // conversion role: 16 rows x 16 segments of 32 k's each
  const int crow = tid >> 4;
  const int cseg = (tid & 15) * 32;

  const int*  xrow = x + (size_t)b * D;
  const float b1c  = b1[n0 + laneN];
  float colsum = 0.f;                   // per-lane column partial over all r

  for (int rt = 0; rt < RT; ++rt) {
    int rg = rt * 16 + crow;
    if (rg > R - 1) rg = R - 1;         // clamp pad rows (excluded from the sum)
    const int* nrow = noise + ((size_t)b * R + rg) * D;

    v8f acc = {};

    // Convert chunk ci of (x XOR noise) into bf16 {0,1} in LDS buffer `buf`.
    auto convert = [&](int ci, int buf) {
      unsigned short (*dst)[ASTR] = tileA[buf];
      const int kc = ci * KC;
#pragma unroll
      for (int j = 0; j < 8; ++j) {
        const int kl = cseg + j * 4;
        const int kg = kc + kl;
        unsigned lo, hi;
        if (kg + 3 < D) {
          // Non-temporal: 631 MB of noise streams once, keep L2 for W1b/x/W2.
          i4 nv = __builtin_nontemporal_load((const i4*)(nrow + kg));
          i4 xv = *(const i4*)(xrow + kg);
          unsigned e0 = ((nv.x ^ xv.x) & 1) ? 0x3F80u : 0u;
          unsigned e1 = ((nv.y ^ xv.y) & 1) ? 0x3F80u : 0u;
          unsigned e2 = ((nv.z ^ xv.z) & 1) ? 0x3F80u : 0u;
          unsigned e3 = ((nv.w ^ xv.w) & 1) ? 0x3F80u : 0u;
          lo = e0 | (e1 << 16);
          hi = e2 | (e3 << 16);
        } else {
          unsigned e[4];
#pragma unroll
          for (int t = 0; t < 4; ++t) {
            int kk = kg + t;
            e[t] = (kk < D) ? ((((nrow[kk]) ^ (xrow[kk])) & 1) ? 0x3F80u : 0u) : 0u;
          }
          lo = e[0] | (e[1] << 16);
          hi = e[2] | (e[3] << 16);
        }
        uint2 p; p.x = lo; p.y = hi;
        *(uint2*)&dst[crow][kl] = p;    // ds_store_b64
      }
    };

    convert(0, 0);
    __syncthreads();

    for (int ci = 0; ci < NCHUNK; ++ci) {
      if (ci + 1 < NCHUNK) convert(ci + 1, (ci + 1) & 1);  // overlap with WMMA below

      const unsigned short (*Ab)[ASTR] = tileA[ci & 1];
      const int kc = ci * KC;
      int kv = D - kc; if (kv > KC) kv = KC;
#pragma unroll 4
      for (int k = 0; k < kv; k += 32) {
        // A fragment (ISA 16-bit 16x32 layout): low half K k..k+7,k+16..23;
        // high half K k+8..15,k+24..31; M = lane&15.
        const unsigned short* ap = &Ab[laneN][k + laneHi * 8];
        v8bf alo = *(const v8bf*)ap;
        v8bf ahi = *(const v8bf*)(ap + 16);
        v16bf afrag = __builtin_shufflevector(
            alo, ahi, 0, 1, 2, 3, 4, 5, 6, 7, 8, 9, 10, 11, 12, 13, 14, 15);

        // B fragment: N = lane&15, K in the same order as the A fragment.
        v16bf bfrag;
#pragma unroll
        for (int s = 0; s < 16; ++s) {
          int kr = kc + k + laneHi * 8 + (s & 7) + (s >> 3) * 16;
          if (kr > D - 1) kr = D - 1;     // A is zero there; clamp keeps loads in-bounds
          bfrag[s] = __builtin_bit_cast(__bf16, W1b[(size_t)kr * H + n0 + laneN]);
        }

        acc = __builtin_amdgcn_wmma_f32_16x16x32_bf16(
            false, afrag, false, bfrag, (short)0, acc, false, false);
      }
      __syncthreads();
    }

    // bias + tanh, accumulate only valid rows (mean over r commutes with GEMM2)
#pragma unroll
    for (int v = 0; v < 8; ++v) {
      int rglob = rt * 16 + v + laneHi * 8;   // C/D layout: M = v + 8*laneHi
      if (rglob < R) colsum += tanhf(acc[v] + b1c);
    }
  }

  hpart[wave][lane] = colsum;
  __syncthreads();
  if (tid < H)
    hsum[tid] = hpart[tid >> 4][tid & 15] + hpart[tid >> 4][(tid & 15) + 16];
  __syncthreads();

  // Tiny GEMM2: out[b,c] = (hsum/R) @ W2 + b2
  if (tid < C) {
    float a2 = 0.f;
#pragma unroll 8
    for (int hh = 0; hh < H; ++hh)
      a2 = fmaf(hsum[hh], W2[hh * C + tid], a2);
    out[b * C + tid] = a2 * (1.0f / R) + b2[tid];
  }
}

extern "C" void kernel_launch(void* const* d_in, const int* in_sizes, int n_in,
                              void* d_out, int out_size, void* d_ws, size_t ws_size,
                              hipStream_t stream) {
  const int*   x     = (const int*)d_in[0];
  const int*   noise = (const int*)d_in[1];
  const float* W1    = (const float*)d_in[2];
  const float* b1    = (const float*)d_in[3];
  const float* W2    = (const float*)d_in[4];
  const float* b2    = (const float*)d_in[5];
  float* out = (float*)d_out;

  unsigned short* W1b = (unsigned short*)d_ws;   // D*H*2 = ~1.58 MB of scratch
  const int n = D * H;
  w1_to_bf16<<<(n + 255) / 256, 256, 0, stream>>>(W1, W1b, n);
  noisy_mlp<<<B, 256, 0, stream>>>(x, noise, W1b, b1, W2, b2, out);
}
